// MultiHeadAttention_70497593196680
// MI455X (gfx1250) — matrix operand order repaired and checked by reading, not compiled, not run on hardware
//
#include <hip/hip_runtime.h>
#include <stdint.h>

#define BS    4
#define SEQ   2048
#define DIN   1024
#define DOUT  1024
#define NHEAD 16
#define HDIM  64

typedef __attribute__((ext_vector_type(16))) __bf16 v16bf;
typedef __attribute__((ext_vector_type(8)))  float  v8f;
typedef unsigned int u32x4 __attribute__((ext_vector_type(4)));
typedef int          i32x4 __attribute__((ext_vector_type(4)));
typedef int          i32x8 __attribute__((ext_vector_type(8)));

#if __has_include(<hip/amd_detail/amd_gfx1250_TDM.h>)
#define TDM_6ARG 1
#endif

union BF16Frag {
    v16bf v;
    uint4 q[2];
    unsigned short u[16];
};

__device__ __forceinline__ unsigned short f2bf(float f) {
    unsigned int u = __float_as_uint(f);
    u += 0x7FFFu + ((u >> 16) & 1u);   // round-to-nearest-even
    return (unsigned short)(u >> 16);
}

// Low 32 bits of a flat shared-aperture address == LDS byte address (ISA 10.2).
__device__ __forceinline__ unsigned lds_addr_of(const void* p) {
    return (unsigned)(unsigned long long)p;
}

// ---- Tensor Data Mover: 2-D bf16 tile load Global->LDS (D# per ISA 8.3/8.4)
//  tile: tile_d0 (contiguous, elements) x tile_d1 (rows); row stride = stride_d0 elems
__device__ __forceinline__ void tdm_load_tile_2d(unsigned lds_addr, const void* gptr,
                                                 unsigned tensor_d0, unsigned tensor_d1,
                                                 unsigned tile_d0, unsigned tile_d1,
                                                 unsigned stride_d0) {
    unsigned long long ga = (unsigned long long)gptr;
    u32x4 g0;
    g0[0] = 1u;                                          // count=1, user mode
    g0[1] = lds_addr;                                    // lds_addr [63:32]
    g0[2] = (unsigned)(ga & 0xFFFFFFFFu);                // global_addr lo
    g0[3] = (unsigned)((ga >> 32) & 0x1FFFFFFu) | (2u << 30);  // addr hi | type=2
    i32x8 g1;
    g1[0] = (int)(1u << 16);                             // data_size=1 (2 bytes)
    g1[1] = (int)((tensor_d0 & 0xFFFFu) << 16);          // tensor_dim0 [79:48]
    g1[2] = (int)((tensor_d0 >> 16) | ((tensor_d1 & 0xFFFFu) << 16)); // dim0 hi | dim1 lo
    g1[3] = (int)((tensor_d1 >> 16) | (tile_d0 << 16));  // dim1 hi | tile_dim0
    g1[4] = (int)(tile_d1 & 0xFFFFu);                    // tile_dim1 (tile_dim2=0)
    g1[5] = (int)stride_d0;                              // tensor_dim0_stride lo
    g1[6] = 0;
    g1[7] = 0;
    i32x4 z4 = {0, 0, 0, 0};
#ifdef TDM_6ARG
    i32x8 z8 = {0, 0, 0, 0, 0, 0, 0, 0};
    __builtin_amdgcn_tensor_load_to_lds(g0, g1, z4, z4, z8, 0);
#else
    __builtin_amdgcn_tensor_load_to_lds(g0, g1, z4, z4, 0);
#endif
}

// ---------------------------------------------------------------- conversions
__global__ void k_f32_to_bf16(const float* __restrict__ in,
                              unsigned short* __restrict__ out, int n) {
    int i = blockIdx.x * blockDim.x + threadIdx.x;
    if (i < n) out[i] = f2bf(in[i]);
}

// in: [K,N] fp32 row-major  ->  out: [N,K] bf16 row-major (transposed weights)
__global__ void k_transpose_bf16(const float* __restrict__ in,
                                 unsigned short* __restrict__ out, int K, int N) {
    int i = blockIdx.x * blockDim.x + threadIdx.x;
    if (i < K * N) {
        int n = i / K, k = i % K;
        out[i] = f2bf(in[(size_t)k * N + n]);
    }
}

// ---------------------------------------------------------------- bf16 GEMM
// 256 threads = 8 waves; block tile 128(M) x 64(N); B panel (64 rows x 32 K)
// staged in LDS by the Tensor Data Mover, double-buffered, shared by all waves.
// mode 0: fp32 out row-major [M,N]
// mode 1: bf16 out [BS][NHEAD][SEQ][HDIM]     (Q, K layout)
// mode 2: bf16 out [BS][NHEAD][HDIM][SEQ]     (V transposed layout)
__global__ __launch_bounds__(256)
void k_gemm_bf16(const unsigned short* __restrict__ A,   // [M,K] bf16
                 const unsigned short* __restrict__ Bt,  // [N,K] bf16
                 const float* __restrict__ bias,         // [N]
                 void* __restrict__ out,
                 int M, int N, int K, int mode) {
    __shared__ __align__(16) unsigned short ldsB[2][64 * 32];   // 2 x 4 KB

    const int lane = threadIdx.x & 31;
    const int wave = threadIdx.x >> 5;
    const int r = lane & 15, half = lane >> 4;
    const int n0 = blockIdx.x * 64;
    const int m0 = blockIdx.y * 128 + wave * 16;

    // Prologue: wave 0 launches DMA of first B panel (k = 0..31)
    if (wave == 0) {
        tdm_load_tile_2d(lds_addr_of(&ldsB[0][0]), Bt + (size_t)n0 * K,
                         (unsigned)K, (unsigned)N, 32u, 64u, (unsigned)K);
    }

    v8f acc[4] = {};
    const unsigned short* arow = A + (size_t)(m0 + r) * K;
    BF16Frag a;                                   // per-wave A fragment (k-step 0)
    a.q[0] = *(const uint4*)(arow + half * 8);
    a.q[1] = *(const uint4*)(arow + 16 + half * 8);

    int p = 0;
    for (int k0 = 0; k0 < K; k0 += 32) {
        // software-pipeline the next A fragment
        BF16Frag an;
        if (k0 + 32 < K) {
            an.q[0] = *(const uint4*)(arow + k0 + 32 + half * 8);
            an.q[1] = *(const uint4*)(arow + k0 + 48 + half * 8);
        }
        if (wave == 0) {
            if (k0 + 32 < K) {
                tdm_load_tile_2d(lds_addr_of(&ldsB[p ^ 1][0]),
                                 Bt + (size_t)n0 * K + (k0 + 32),
                                 (unsigned)K, (unsigned)N, 32u, 64u, (unsigned)K);
                __builtin_amdgcn_s_wait_tensorcnt(1);   // current panel done
            } else {
                __builtin_amdgcn_s_wait_tensorcnt(0);
            }
        }
        __syncthreads();                               // publish LDS panel

        const unsigned short* bp = &ldsB[p][0];
        for (int t = 0; t < 4; ++t) {
            const unsigned short* brow = bp + (t * 16 + r) * 32 + half * 8;
            BF16Frag b;
            b.q[0] = *(const uint4*)(brow);
            b.q[1] = *(const uint4*)(brow + 16);
            acc[t] = __builtin_amdgcn_wmma_f32_16x16x32_bf16(
                false, a.v, false, b.v, (short)0, acc[t], false, false);
        }
        __syncthreads();                               // reads done before reuse
        if (k0 + 32 < K) a = an;
        p ^= 1;
    }

    for (int t = 0; t < 4; ++t) {
        int ng = n0 + t * 16 + r;
        float bia = bias[ng];
        for (int i = 0; i < 8; ++i) {
            int mg = m0 + i + half * 8;
            float val = acc[t][i] + bia;
            if (mode == 0) {
                ((float*)out)[(size_t)mg * N + ng] = val;
            } else {
                int bb = mg / SEQ, s = mg % SEQ;
                int hh = ng / HDIM, d = ng % HDIM;
                unsigned short bf = f2bf(val);
                if (mode == 1)
                    ((unsigned short*)out)[(((size_t)(bb * NHEAD + hh) * SEQ) + s) * HDIM + d] = bf;
                else
                    ((unsigned short*)out)[(((size_t)(bb * NHEAD + hh) * HDIM) + d) * SEQ + s] = bf;
            }
        }
    }
}

// ---------------------------------------------------------------- attention
// One wave handles 16 query rows of one (b,h). Flash-style online softmax.
// Q,K: bf16 [BS][NHEAD][SEQ][HDIM];  Vt: bf16 [BS][NHEAD][HDIM][SEQ]
// ctx out: bf16 [BS][SEQ][DOUT]
__global__ __launch_bounds__(32)
void k_attn(const unsigned short* __restrict__ Q,
            const unsigned short* __restrict__ Kmat,
            const unsigned short* __restrict__ Vt,
            unsigned short* __restrict__ ctx) {
    __shared__ __align__(16) unsigned short ldsP[16 * 32];   // P tile, row stride 32

    const int lane = threadIdx.x & 31;
    const int r = lane & 15, half = lane >> 4;
    const int qt = blockIdx.x, h = blockIdx.y, b = blockIdx.z;
    const int q0 = qt * 16;
    const size_t headOff = (size_t)(b * NHEAD + h) * SEQ * HDIM;
    const unsigned short* Qh = Q + headOff;
    const unsigned short* Kh = Kmat + headOff;
    const unsigned short* Vh = Vt + headOff;   // [HDIM][SEQ]

    // Q A-fragments: hd 0..31 (a0), 32..63 (a1)
    const unsigned short* qrow = Qh + (size_t)(q0 + r) * HDIM;
    BF16Frag a0, a1;
    a0.q[0] = *(const uint4*)(qrow + half * 8);
    a0.q[1] = *(const uint4*)(qrow + 16 + half * 8);
    a1.q[0] = *(const uint4*)(qrow + 32 + half * 8);
    a1.q[1] = *(const uint4*)(qrow + 48 + half * 8);

    v8f acc[4] = {};
    float mstat[8], lstat[8];
    for (int i = 0; i < 8; ++i) { mstat[i] = -__builtin_inff(); lstat[i] = 0.f; }
    const float scale = 0.125f;   // 1/sqrt(64)

    for (int j = 0; j < q0 + 16; j += 32) {   // key blocks of 32, causal bound
        // prefetch next key block rows into cache
        if (j + 32 < q0 + 16)
            __builtin_prefetch(Kh + (size_t)(j + 32 + r) * HDIM, 0, 1);

        // ---- scores S = Q K^T for keys [j, j+32)
        v8f s[2];
        for (int f = 0; f < 2; ++f) {
            const unsigned short* krow = Kh + (size_t)(j + f * 16 + r) * HDIM;
            BF16Frag blo, bhi;
            blo.q[0] = *(const uint4*)(krow + half * 8);
            blo.q[1] = *(const uint4*)(krow + 16 + half * 8);
            bhi.q[0] = *(const uint4*)(krow + 32 + half * 8);
            bhi.q[1] = *(const uint4*)(krow + 48 + half * 8);
            v8f sf = {};
            sf = __builtin_amdgcn_wmma_f32_16x16x32_bf16(
                false, a0.v, false, blo.v, (short)0, sf, false, false);
            sf = __builtin_amdgcn_wmma_f32_16x16x32_bf16(
                false, a1.v, false, bhi.v, (short)0, sf, false, false);
            s[f] = sf;
        }

        // ---- causal mask, scale, online softmax update
        float p0[8], p1[8];
        for (int i = 0; i < 8; ++i) {
            int qg = q0 + i + half * 8;
            int key0 = j + r, key1 = j + 16 + r;
            float v0 = (key0 <= qg) ? s[0][i] * scale : -__builtin_inff();
            float v1 = (key1 <= qg) ? s[1][i] * scale : -__builtin_inff();
            float t = fmaxf(v0, v1);
            for (int off = 1; off <= 8; off <<= 1)
                t = fmaxf(t, __shfl_xor(t, off, 32));   // reduce within 16-lane half
            float mnew = fmaxf(mstat[i], t);
            float alpha = __expf(mstat[i] - mnew);      // exp(-inf)=0 on first block
            float e0 = __expf(v0 - mnew);
            float e1 = __expf(v1 - mnew);
            float rs = e0 + e1;
            for (int off = 1; off <= 8; off <<= 1)
                rs += __shfl_xor(rs, off, 32);
            lstat[i] = lstat[i] * alpha + rs;
            mstat[i] = mnew;
            for (int t2 = 0; t2 < 4; ++t2) acc[t2][i] = acc[t2][i] * alpha;
            p0[i] = e0; p1[i] = e1;
        }

        // ---- P: C-layout -> A-layout via LDS (bf16, row-major 16x32 tile)
        for (int i = 0; i < 8; ++i) {
            ldsP[(i + half * 8) * 32 + r]      = f2bf(p0[i]);
            ldsP[(i + half * 8) * 32 + 16 + r] = f2bf(p1[i]);
        }
        __syncthreads();
        // A-fragment = two contiguous 8-elem runs per lane -> two ds_load_b128
        BF16Frag ap;
        ap.q[0] = *(const uint4*)(&ldsP[r * 32 + half * 8]);        // K=half*8+0..7
        ap.q[1] = *(const uint4*)(&ldsP[r * 32 + 16 + half * 8]);   // K=16+half*8+0..7
        __syncthreads();

        // ---- O += P @ V   (V transposed: contiguous B-fragment loads)
        for (int t2 = 0; t2 < 4; ++t2) {
            const unsigned short* vrow =
                Vh + (size_t)(t2 * 16 + r) * SEQ + j + half * 8;
            BF16Frag bvf;
            bvf.q[0] = *(const uint4*)(vrow);
            bvf.q[1] = *(const uint4*)(vrow + 16);
            acc[t2] = __builtin_amdgcn_wmma_f32_16x16x32_bf16(
                false, ap.v, false, bvf.v, (short)0, acc[t2], false, false);
        }
    }

    // ---- normalize and write ctx (bf16, [BS][SEQ][DOUT])
    for (int t2 = 0; t2 < 4; ++t2) {
        for (int i = 0; i < 8; ++i) {
            int s = q0 + i + half * 8;
            int col = h * HDIM + t2 * 16 + r;
            float o = acc[t2][i] / lstat[i];
            ctx[((size_t)b * SEQ + s) * DOUT + col] = f2bf(o);
        }
    }
}

// ---------------------------------------------------------------- host
extern "C" void kernel_launch(void* const* d_in, const int* in_sizes, int n_in,
                              void* d_out, int out_size, void* d_ws, size_t ws_size,
                              hipStream_t stream) {
    const float* x   = (const float*)d_in[0];
    const float* Wq  = (const float*)d_in[1];
    const float* bq  = (const float*)d_in[2];
    const float* Wk  = (const float*)d_in[3];
    const float* bk  = (const float*)d_in[4];
    const float* Wv  = (const float*)d_in[5];
    const float* bvv = (const float*)d_in[6];
    const float* Wo  = (const float*)d_in[7];
    const float* bo  = (const float*)d_in[8];
    float* out = (float*)d_out;

    const size_t M = (size_t)BS * SEQ;   // 8192

    char* ws = (char*)d_ws;
    size_t off = 0;
    auto alloc = [&](size_t bytes) -> char* {
        char* p = ws + off;
        off += (bytes + 255) & ~(size_t)255;
        return p;
    };
    unsigned short* x_bf = (unsigned short*)alloc(M * DIN * 2);            // 16 MB
    unsigned short* wq_t = (unsigned short*)alloc((size_t)DIN * DOUT * 2); // 2 MB
    unsigned short* wk_t = (unsigned short*)alloc((size_t)DIN * DOUT * 2);
    unsigned short* wv_t = (unsigned short*)alloc((size_t)DIN * DOUT * 2);
    unsigned short* wo_t = (unsigned short*)alloc((size_t)DOUT * DOUT * 2);
    unsigned short* qb   = (unsigned short*)alloc(M * DOUT * 2);           // 16 MB
    unsigned short* kb   = (unsigned short*)alloc(M * DOUT * 2);
    unsigned short* vtb  = (unsigned short*)alloc(M * DOUT * 2);
    unsigned short* ctxb = (unsigned short*)alloc(M * DOUT * 2);

    int nconv = (int)(M * DIN);
    k_f32_to_bf16<<<(nconv + 255) / 256, 256, 0, stream>>>(x, x_bf, nconv);

    int nw = DIN * DOUT;
    k_transpose_bf16<<<(nw + 255) / 256, 256, 0, stream>>>(Wq, wq_t, DIN, DOUT);
    k_transpose_bf16<<<(nw + 255) / 256, 256, 0, stream>>>(Wk, wk_t, DIN, DOUT);
    k_transpose_bf16<<<(nw + 255) / 256, 256, 0, stream>>>(Wv, wv_t, DIN, DOUT);
    k_transpose_bf16<<<(nw + 255) / 256, 256, 0, stream>>>(Wo, wo_t, DOUT, DOUT);

    dim3 g1(DOUT / 64, (unsigned)(M / 128));
    k_gemm_bf16<<<g1, 256, 0, stream>>>(x_bf, wq_t, bq,  qb,  (int)M, DOUT, DIN, 1);
    k_gemm_bf16<<<g1, 256, 0, stream>>>(x_bf, wk_t, bk,  kb,  (int)M, DOUT, DIN, 1);
    k_gemm_bf16<<<g1, 256, 0, stream>>>(x_bf, wv_t, bvv, vtb, (int)M, DOUT, DIN, 2);

    dim3 ga(SEQ / 16, NHEAD, BS);
    k_attn<<<ga, 32, 0, stream>>>(qb, kb, vtb, ctxb);

    k_gemm_bf16<<<g1, 256, 0, stream>>>(ctxb, wo_t, bo, out, (int)M, DOUT, DIN, 0);
}
